// LSTM_43645457662714
// MI455X (gfx1250) — matrix operand order repaired and checked by reading, not compiled, not run on hardware
//
#include <hip/hip_runtime.h>

// ---------------------------------------------------------------------------
// 2-layer LSTM (B=512, T=200, H=512) for gfx1250.
//  * bf16 WMMA (v_wmma_f32_16x16x32_bf16), fp32 accumulate, fused gate epilogue
//  * Weight panels staged into LDS once per block via the Tensor Data Mover
//    (tensor_load_to_lds + s_wait_tensorcnt), B fragments via ds_load_b128
//  * Software-pipelined A-fragment loads (prefetch next K-chunk), B fragments
//    hoisted so one DS wait covers a burst of 8 WMMAs
//  * Each wave: 32 batch rows x 16 units (2 row-tiles share every B fragment)
// ---------------------------------------------------------------------------

#define BB 512
#define TT 200
#define HH 512
#define GG 2048  // 4*H

typedef __attribute__((ext_vector_type(8)))  __bf16 v8bf;
typedef __attribute__((ext_vector_type(16))) __bf16 v16bf;
typedef __attribute__((ext_vector_type(8)))  float  v8f;
typedef unsigned int v4u __attribute__((ext_vector_type(4)));
typedef int          v4i __attribute__((ext_vector_type(4)));
typedef int          v8i __attribute__((ext_vector_type(8)));

// ---- WMMA helpers ---------------------------------------------------------

__device__ __forceinline__ v8f wmma_bf16(v16bf a, v16bf b, v8f c) {
    return __builtin_amdgcn_wmma_f32_16x16x32_bf16(
        false, a, false, b, (short)0, c, false, false);
}

__device__ __forceinline__ v16bf make16(v8bf lo, v8bf hi8) {
    return __builtin_shufflevector(lo, hi8, 0, 1, 2, 3, 4, 5, 6, 7,
                                   8, 9, 10, 11, 12, 13, 14, 15);
}

// A fragment (16x32 bf16), row-major [rows][lda].
// lane<16 (row M0+ml): K = k0+0..7 and k0+16..23 ; lane>=16: +8 / +24.
__device__ __forceinline__ v16bf load_a_frag(const __bf16* __restrict__ base,
                                             size_t lda, int M0, int k0,
                                             int ml, int hi) {
    const __bf16* p = base + (size_t)(M0 + ml) * lda + (size_t)k0 + hi * 8;
    return make16(*(const v8bf*)(p), *(const v8bf*)(p + 16));
}

// B fragment (32x16 bf16) from an LDS panel row (K contiguous):
// lane column N = ml; lane<16 holds K=k0+0..7 and k0+16..23, lane>=16 +8 / +24.
__device__ __forceinline__ v16bf load_b_lds(const __bf16* __restrict__ prow,
                                            int k0, int hi) {
    const __bf16* p = prow + (size_t)k0 + hi * 8;
    return make16(*(const v8bf*)(p), *(const v8bf*)(p + 16));
}

__device__ __forceinline__ float sigmoidf_(float x) {
    return 1.0f / (1.0f + __expf(-x));
}

// ---- Tensor Data Mover: load one 4-gate weight panel into LDS -------------
// Panel in LDS: [gate 0..3][col 0..15][K tile_u*4 bf16], contiguous.
// Source: gate-major WT[4H][H]; tile start = WT + u0*H (+ k-offset), with
//   X = tile_u 8-byte units (K-contiguous), Y = 16 cols (stride 128 units),
//   Z = 4 gates (stride H rows = 65536 units).

__device__ __forceinline__ void tdm_call(v4u g0, v8i g1, v4i g2, v4i g3) {
#if defined(__clang_major__) && (__clang_major__ >= 23)
    v8i g4 = {0, 0, 0, 0, 0, 0, 0, 0};
    __builtin_amdgcn_tensor_load_to_lds(g0, g1, g2, g3, g4, 0);
#else
    __builtin_amdgcn_tensor_load_to_lds(g0, g1, g2, g3, 0);
#endif
}

__device__ __forceinline__ void tdm_load_panel(const void* gbase,
                                               unsigned lds_off_bytes,
                                               unsigned tile_u /* 8B units */) {
    unsigned long long ga = (unsigned long long)gbase;
    v4u g0 = {1u,                                  // count=1, user descriptor
              lds_off_bytes,                       // lds_addr (dyn-LDS base = 0)
              (unsigned)ga,                        // global_addr[31:0]
              (unsigned)((ga >> 32) & 0x01FFFFFFull) | 0x80000000u};  // type=2
    const unsigned s0 = 128u;    // row stride: 512 bf16 = 128 x 8B
    const unsigned s1 = 65536u;  // gate stride: H rows = 512*1024B / 8
    v8i g1 = {
        (int)(3u << 16),                                  // data_size = 8 bytes
        (int)((tile_u & 0xFFFFu) << 16),                  // tensor_dim0[15:0]
        (int)(((tile_u >> 16) & 0xFFFFu) | (16u << 16)),  // dim0[31:16] | dim1[15:0]
        (int)((tile_u & 0xFFFFu) << 16),                  // dim1[31:16]=0 | tile_dim0
        (int)(16u | (4u << 16)),                          // tile_dim1=16 | tile_dim2=4
        (int)s0,                                          // tensor_dim0_stride[31:0]
        (int)((s1 & 0xFFFFu) << 16),                      // s0[47:32]=0 | s1[15:0]
        (int)(s1 >> 16)                                   // s1[47:16]
    };
    v4i g2 = {4, 0, 0, 0};  // tensor_dim2 = 4 gates
    v4i g3 = {0, 0, 0, 0};
    tdm_call(g0, g1, g2, g3);
}

// ---- prep kernels ---------------------------------------------------------

// src fp32 [K][G] row-major -> dst bf16 [G][K] (gate-major, K contiguous)
__global__ void convert_transpose(const float* __restrict__ src,
                                  __bf16* __restrict__ dst, int K, int G) {
    int i = blockIdx.x * blockDim.x + threadIdx.x;
    if (i >= K * G) return;
    int k = i / G;
    int g = i % G;
    dst[(size_t)g * K + k] = (__bf16)src[i];
}

__global__ void init_state(__bf16* __restrict__ h0, float* __restrict__ c) {
    int i = blockIdx.x * blockDim.x + threadIdx.x;
    if (i >= BB * HH) return;
    h0[i] = (__bf16)0.0f;
    c[i] = 0.0f;
}

// ---- gate epilogues -------------------------------------------------------

__device__ __forceinline__ void gates1(const v8f* acc, int M0, int hi, int u, int t,
                                       const float* __restrict__ inp,
                                       float wxi, float wxf, float wxg, float wxo,
                                       float bi, float bf, float bg, float bo,
                                       float* __restrict__ c,
                                       __bf16* __restrict__ hnext,
                                       __bf16* __restrict__ seq1) {
#pragma unroll
    for (int r = 0; r < 8; ++r) {
        const int row = M0 + r + 8 * hi;
        const float xv = inp[(size_t)row * TT + t];
        const float ig = sigmoidf_(acc[0][r] + xv * wxi + bi);
        const float fg = sigmoidf_(acc[1][r] + xv * wxf + bf);
        const float gg = tanhf(acc[2][r] + xv * wxg + bg);
        const float og = sigmoidf_(acc[3][r] + xv * wxo + bo);
        const size_t cu = (size_t)row * HH + u;
        const float cn = fg * c[cu] + ig * gg;
        c[cu] = cn;
        const float hn = og * tanhf(cn);
        hnext[cu] = (__bf16)hn;
        seq1[((size_t)row * TT + t) * HH + u] = (__bf16)hn;
    }
}

__device__ __forceinline__ void gates2(const v8f* acc, int M0, int hi, int u,
                                       float bi, float bf, float bg, float bo,
                                       float* __restrict__ c,
                                       __bf16* __restrict__ hnext) {
#pragma unroll
    for (int r = 0; r < 8; ++r) {
        const int row = M0 + r + 8 * hi;
        const float ig = sigmoidf_(acc[0][r] + bi);
        const float fg = sigmoidf_(acc[1][r] + bf);
        const float gg = tanhf(acc[2][r] + bg);
        const float og = sigmoidf_(acc[3][r] + bo);
        const size_t cu = (size_t)row * HH + u;
        const float cn = fg * c[cu] + ig * gg;
        c[cu] = cn;
        hnext[cu] = (__bf16)(og * tanhf(cn));
    }
}

// ---- LSTM step kernels ----------------------------------------------------
// 64 blocks x 256 threads. blockIdx = unittile*2 + rowgroup.
// Wave w handles rows M0..M0+31 (two 16-row tiles) for units u0..u0+15.

__global__ void __launch_bounds__(256)
lstm_step1(const float* __restrict__ inp, const float* __restrict__ Wx1,
           const float* __restrict__ b1, const __bf16* __restrict__ Wh1T,
           const __bf16* __restrict__ hprev, __bf16* __restrict__ hnext,
           float* __restrict__ c, __bf16* __restrict__ seq1, int t) {
    extern __shared__ char smem_raw[];
    __bf16* Bpan = (__bf16*)smem_raw;  // [4][16][512] bf16 = 64 KB

    const int u0 = (blockIdx.x >> 1) * 16;
    const int rowgrp = blockIdx.x & 1;
    const int wave = threadIdx.x >> 5;
    const int lane = threadIdx.x & 31;
    const int ml = lane & 15;
    const int hi = lane >> 4;

    if (threadIdx.x < 32) {  // one wave issues the TDM panel fill
        tdm_load_panel(Wh1T + (size_t)u0 * HH, 0u, 128u /* full 512-elem rows */);
        __builtin_amdgcn_s_wait_tensorcnt(0);
    }
    __syncthreads();

    const int M0 = rowgrp * 256 + wave * 32;
    const __bf16* brow = Bpan + (size_t)ml * HH;  // + g*16*HH, then k0

    v8f acc0[4] = {}, acc1[4] = {};

    // software pipeline: A fragments for chunk k prefetched during chunk k-32
    v16bf a0 = load_a_frag(hprev, HH, M0, 0, ml, hi);
    v16bf a1 = load_a_frag(hprev, HH, M0 + 16, 0, ml, hi);

#pragma unroll
    for (int k0 = 0; k0 < HH; k0 += 32) {
        v16bf na0, na1;
        if (k0 + 32 < HH) {
            na0 = load_a_frag(hprev, HH, M0, k0 + 32, ml, hi);
            na1 = load_a_frag(hprev, HH, M0 + 16, k0 + 32, ml, hi);
        }
        v16bf b[4];
#pragma unroll
        for (int g = 0; g < 4; ++g)
            b[g] = load_b_lds(brow + (size_t)g * 16 * HH, k0, hi);
#pragma unroll
        for (int g = 0; g < 4; ++g) {
            acc0[g] = wmma_bf16(a0, b[g], acc0[g]);
            acc1[g] = wmma_bf16(a1, b[g], acc1[g]);
        }
        a0 = na0;
        a1 = na1;
    }

    const int u = u0 + ml;
    const float bi = b1[u], bf = b1[HH + u], bg = b1[2 * HH + u], bo = b1[3 * HH + u];
    const float wxi = Wx1[u], wxf = Wx1[HH + u], wxg = Wx1[2 * HH + u], wxo = Wx1[3 * HH + u];
    gates1(acc0, M0,      hi, u, t, inp, wxi, wxf, wxg, wxo, bi, bf, bg, bo, c, hnext, seq1);
    gates1(acc1, M0 + 16, hi, u, t, inp, wxi, wxf, wxg, wxo, bi, bf, bg, bo, c, hnext, seq1);
}

__global__ void __launch_bounds__(256)
lstm_step2(const __bf16* __restrict__ seq1, const __bf16* __restrict__ Wx2T,
           const float* __restrict__ b2, const __bf16* __restrict__ Wh2T,
           const __bf16* __restrict__ hprev, __bf16* __restrict__ hnext,
           float* __restrict__ c, int t) {
    extern __shared__ char smem_raw[];
    __bf16* Bpan = (__bf16*)smem_raw;  // [2][4][16][256] bf16 = 64 KB (K-half)

    const int u0 = (blockIdx.x >> 1) * 16;
    const int rowgrp = blockIdx.x & 1;
    const int wave = threadIdx.x >> 5;
    const int lane = threadIdx.x & 31;
    const int ml = lane & 15;
    const int hi = lane >> 4;
    const int M0 = rowgrp * 256 + wave * 32;

    const __bf16* xbase = seq1 + (size_t)t * HH;  // row b at b*T*H + t*H
    const size_t xlda = (size_t)TT * HH;

    v8f acc0[4] = {}, acc1[4] = {};

    for (int kh = 0; kh < 2; ++kh) {
        __syncthreads();  // previous half's panel fully consumed
        if (threadIdx.x < 32) {
            const size_t koff = (size_t)kh * 256;
            tdm_load_panel(Wx2T + (size_t)u0 * HH + koff, 0u,     64u /* 256-elem rows */);
            tdm_load_panel(Wh2T + (size_t)u0 * HH + koff, 32768u, 64u);
            __builtin_amdgcn_s_wait_tensorcnt(0);
        }
        __syncthreads();

        const int kbase = kh * 256;
        const __bf16* browx = Bpan + (size_t)ml * 256;       // mat 0 (Wx2)
        const __bf16* browh = browx + (size_t)4 * 16 * 256;  // mat 1 (Wh2)

        // software pipeline across the 8 chunks of this K-half
        v16bf ax0 = load_a_frag(xbase, xlda, M0, kbase, ml, hi);
        v16bf ax1 = load_a_frag(xbase, xlda, M0 + 16, kbase, ml, hi);
        v16bf ah0 = load_a_frag(hprev, HH, M0, kbase, ml, hi);
        v16bf ah1 = load_a_frag(hprev, HH, M0 + 16, kbase, ml, hi);

#pragma unroll
        for (int kk = 0; kk < 256; kk += 32) {
            v16bf nx0, nx1, nh0, nh1;
            if (kk + 32 < 256) {
                const int kn = kbase + kk + 32;
                nx0 = load_a_frag(xbase, xlda, M0, kn, ml, hi);
                nx1 = load_a_frag(xbase, xlda, M0 + 16, kn, ml, hi);
                nh0 = load_a_frag(hprev, HH, M0, kn, ml, hi);
                nh1 = load_a_frag(hprev, HH, M0 + 16, kn, ml, hi);
            }
            v16bf bx[4];
#pragma unroll
            for (int g = 0; g < 4; ++g)
                bx[g] = load_b_lds(browx + (size_t)g * 16 * 256, kk, hi);
#pragma unroll
            for (int g = 0; g < 4; ++g) {
                acc0[g] = wmma_bf16(ax0, bx[g], acc0[g]);
                acc1[g] = wmma_bf16(ax1, bx[g], acc1[g]);
            }
            v16bf bh[4];
#pragma unroll
            for (int g = 0; g < 4; ++g)
                bh[g] = load_b_lds(browh + (size_t)g * 16 * 256, kk, hi);
#pragma unroll
            for (int g = 0; g < 4; ++g) {
                acc0[g] = wmma_bf16(ah0, bh[g], acc0[g]);
                acc1[g] = wmma_bf16(ah1, bh[g], acc1[g]);
            }
            ax0 = nx0; ax1 = nx1; ah0 = nh0; ah1 = nh1;
        }
    }

    const int u = u0 + ml;
    const float bi = b2[u], bf = b2[HH + u], bg = b2[2 * HH + u], bo = b2[3 * HH + u];
    gates2(acc0, M0,      hi, u, bi, bf, bg, bo, c, hnext);
    gates2(acc1, M0 + 16, hi, u, bi, bf, bg, bo, c, hnext);
}

// ---- final dense ----------------------------------------------------------

__global__ void fc_kernel(const __bf16* __restrict__ h, const float* __restrict__ Wfc,
                          const float* __restrict__ bfc, float* __restrict__ out) {
    int b = blockIdx.x * blockDim.x + threadIdx.x;
    if (b >= BB) return;
    float s = 0.0f;
    const __bf16* hr = h + (size_t)b * HH;
    for (int k = 0; k < HH; ++k) s += (float)hr[k] * Wfc[k];
    out[b] = s + bfc[0];
}

// ---- host launcher --------------------------------------------------------

extern "C" void kernel_launch(void* const* d_in, const int* in_sizes, int n_in,
                              void* d_out, int out_size, void* d_ws, size_t ws_size,
                              hipStream_t stream) {
    const float* inp = (const float*)d_in[0];
    const float* Wx1 = (const float*)d_in[1];
    const float* Wh1 = (const float*)d_in[2];
    const float* b1  = (const float*)d_in[3];
    const float* Wx2 = (const float*)d_in[4];
    const float* Wh2 = (const float*)d_in[5];
    const float* b2  = (const float*)d_in[6];
    const float* Wfc = (const float*)d_in[7];
    const float* bfc = (const float*)d_in[8];
    float* out = (float*)d_out;

    char* ws = (char*)d_ws;
    size_t off = 0;
    auto take = [&](size_t bytes) { char* p = ws + off; off += bytes; return p; };
    __bf16* Wh1T  = (__bf16*)take((size_t)GG * HH * 2);       // 2 MB
    __bf16* Wx2T  = (__bf16*)take((size_t)GG * HH * 2);       // 2 MB
    __bf16* Wh2T  = (__bf16*)take((size_t)GG * HH * 2);       // 2 MB
    __bf16* hbuf0 = (__bf16*)take((size_t)BB * HH * 2);       // 512 KB
    __bf16* hbuf1 = (__bf16*)take((size_t)BB * HH * 2);       // 512 KB
    float*  cbuf  = (float*)take((size_t)BB * HH * 4);        // 1 MB
    __bf16* seq1  = (__bf16*)take((size_t)BB * TT * HH * 2);  // ~105 MB
    (void)ws_size; (void)in_sizes; (void)n_in; (void)out_size;

    const int convN = HH * GG;
    convert_transpose<<<(convN + 255) / 256, 256, 0, stream>>>(Wh1, Wh1T, HH, GG);
    convert_transpose<<<(convN + 255) / 256, 256, 0, stream>>>(Wx2, Wx2T, HH, GG);
    convert_transpose<<<(convN + 255) / 256, 256, 0, stream>>>(Wh2, Wh2T, HH, GG);
    init_state<<<(BB * HH + 255) / 256, 256, 0, stream>>>(hbuf0, cbuf);

    // layer 1: read h[t%2], write h[(t+1)%2]; T even -> final state in hbuf0
    for (int t = 0; t < TT; ++t) {
        __bf16* hp = (t & 1) ? hbuf1 : hbuf0;
        __bf16* hn = (t & 1) ? hbuf0 : hbuf1;
        lstm_step1<<<64, 256, 65536, stream>>>(inp, Wx1, b1, Wh1T, hp, hn, cbuf, seq1, t);
    }
    // layer 2: seeded by layer-1 final (h,c) by continuing on same state buffers
    for (int t = 0; t < TT; ++t) {
        __bf16* hp = (t & 1) ? hbuf1 : hbuf0;
        __bf16* hn = (t & 1) ? hbuf0 : hbuf1;
        lstm_step2<<<64, 256, 65536, stream>>>(seq1, Wx2T, b2, Wh2T, hp, hn, cbuf, t);
    }
    fc_kernel<<<(BB + 255) / 256, 256, 0, stream>>>(hbuf0, Wfc, bfc, out);
}
